// GAT_graph_branch_66864050864635
// MI455X (gfx1250) — matrix operand order repaired and checked by reading, not compiled, not run on hardware
//
#include <hip/hip_runtime.h>
#include <math.h>

// ---------------- problem constants (from reference) ----------------
#define NNODES 50000
#define NEDGES 800000
#define ETOT   (NEDGES + NNODES)   // edges + self loops
#define FIN    128
#define FE     16
#define CCH    64                  // out channels per head
#define NH1    2                   // heads, block 1
#define GG     8                   // graphs
#define F1     (NH1 * CCH)         // 128
#define F2     (CCH)               // 64
#define BN_EPS 1e-5f
#define SLOPE  0.2f

typedef __attribute__((ext_vector_type(16))) _Float16 v16h;
typedef __attribute__((ext_vector_type(8)))  float    v8f;

// ---------------- workspace layout (floats) ----------------
constexpr size_t O_DEG   = 0;
constexpr size_t O_LOOP  = O_DEG   + NNODES;                    // self-loop attr sums -> means
constexpr size_t O_XP1   = O_LOOP  + (size_t)NNODES * FE;
constexpr size_t O_XL1   = O_XP1   + (size_t)NNODES * F1;
constexpr size_t O_XR1   = O_XL1   + (size_t)NNODES * F1;
constexpr size_t O_OUT1  = O_XR1   + (size_t)NNODES * F1;       // atomic accum
constexpr size_t O_H1    = O_OUT1  + (size_t)NNODES * F1;
constexpr size_t O_LOG1  = O_H1    + (size_t)NNODES * F1;       // logits -> num
constexpr size_t O_LMAX1 = O_LOG1  + (size_t)ETOT   * NH1;      // encoded u32
constexpr size_t O_DEN1  = O_LMAX1 + (size_t)NNODES * NH1;
constexpr size_t O_BS1   = O_DEN1  + (size_t)NNODES * NH1;
constexpr size_t O_BQ1   = O_BS1   + F1;
constexpr size_t O_XP2   = O_BQ1   + F1;
constexpr size_t O_XL2   = O_XP2   + (size_t)NNODES * F2;
constexpr size_t O_XR2   = O_XL2   + (size_t)NNODES * F2;
constexpr size_t O_OUT2  = O_XR2   + (size_t)NNODES * F2;
constexpr size_t O_H2    = O_OUT2  + (size_t)NNODES * F2;
constexpr size_t O_LOG2  = O_H2    + (size_t)NNODES * F2;
constexpr size_t O_LMAX2 = O_LOG2  + (size_t)ETOT;
constexpr size_t O_DEN2  = O_LMAX2 + NNODES;
constexpr size_t O_BS2   = O_DEN2  + NNODES;
constexpr size_t O_BQ2   = O_BS2   + F2;
constexpr size_t O_PSUM  = O_BQ2   + F2;
constexpr size_t O_PMAX  = O_PSUM  + (size_t)GG * F2;
constexpr size_t O_PCNT  = O_PMAX  + (size_t)GG * F2;
constexpr size_t WS_FLOATS = O_PCNT + GG;

// ---------------- helpers ----------------
__device__ __forceinline__ unsigned int fenc(float f) {
  unsigned int u = __float_as_uint(f);
  return (u & 0x80000000u) ? ~u : (u | 0x80000000u);
}
__device__ __forceinline__ float fdec(unsigned int u) {
  unsigned int b = (u & 0x80000000u) ? (u & 0x7FFFFFFFu) : ~u;
  return __uint_as_float(b);
}

__global__ void fill_zero_kernel(float* p, size_t n) {
  size_t i = (size_t)blockIdx.x * blockDim.x + threadIdx.x;
  size_t stride = (size_t)gridDim.x * blockDim.x;
  for (; i < n; i += stride) p[i] = 0.0f;
}

// ---------------- self loop attr: mean of incoming edge_attr per dst ----------------
// thread per (edge, attr-component): coalesced reads, lane-spread atomics
__global__ void loop_accum_kernel(const int* __restrict__ ei, const float* __restrict__ ea,
                                  float* __restrict__ deg, float* __restrict__ loopsum) {
  long long idx = (long long)blockIdx.x * blockDim.x + threadIdx.x;
  if (idx >= (long long)NEDGES * FE) return;
  int e = (int)(idx >> 4);
  int j = (int)(idx & 15);
  int d = ei[NEDGES + e];
  atomicAdd(&loopsum[(size_t)d * FE + j], ea[idx]);
  if (j == 0) atomicAdd(&deg[d], 1.0f);
}
__global__ void loop_final_kernel(const float* __restrict__ deg, float* __restrict__ loopsum) {
  int idx = blockIdx.x * blockDim.x + threadIdx.x;
  if (idx >= NNODES * FE) return;
  loopsum[idx] /= fmaxf(deg[idx / FE], 1.0f);
}

// ---------------- WMMA GEMM:  Y[M,Nc] = X[M,K] * W[K,Nc] + bias ----------------
// wave computes a 16x64 output tile: 4 WMMA accumulators share one A fragment.
// f32 data converted to f16 fragments, f32 accumulate. Nc must be multiple of 64.
__global__ void gemm_wmma_kernel(const float* __restrict__ X, const float* __restrict__ W,
                                 const float* __restrict__ bias, float* __restrict__ Y,
                                 int M, int K, int Nc) {
  int wave = (int)((blockIdx.x * blockDim.x + threadIdx.x) >> 5);
  int lane = threadIdx.x & 31;
  int ntn = Nc >> 6;                     // 64-wide column groups
  int ntiles = (M >> 4) * ntn;
  if (wave >= ntiles) return;
  int tm = wave / ntn, tn = wave % ntn;
  int m0 = tm << 4, n0 = tn << 6;
  int g  = lane >> 4;                    // lane group 0/1
  int lm = lane & 15;

  v8f acc0 = {}, acc1 = {}, acc2 = {}, acc3 = {};
  for (int k0 = 0; k0 < K; k0 += 32) {
    // A fragment: lane (g,lm) holds row m0+lm, K chunks [k0+8g..+7] and [k0+16+8g..+7]
    v16h a;
    const float* xrow = X + (size_t)(m0 + lm) * K + k0 + 8 * g;
    #pragma unroll
    for (int e = 0; e < 8; ++e)  a[e] = (_Float16)xrow[e];
    #pragma unroll
    for (int e = 8; e < 16; ++e) a[e] = (_Float16)xrow[e + 8];
    // B fragments: lane (g,lm) holds col n0+16t+lm, K rows [k0+8g..+7] and [k0+16+8g..+7] -> 4 tiles
    const float* wrow = W + (size_t)(k0 + 8 * g) * Nc + n0 + lm;
    v16h b0, b1, b2, b3;
    #pragma unroll
    for (int e = 0; e < 16; ++e) {
      size_t kk = (size_t)(e < 8 ? e : e + 8) * Nc;
      b0[e] = (_Float16)wrow[kk];
      b1[e] = (_Float16)wrow[kk + 16];
      b2[e] = (_Float16)wrow[kk + 32];
      b3[e] = (_Float16)wrow[kk + 48];
    }
    acc0 = __builtin_amdgcn_wmma_f32_16x16x32_f16(false, a, false, b0, (short)0, acc0, false, false);
    acc1 = __builtin_amdgcn_wmma_f32_16x16x32_f16(false, a, false, b1, (short)0, acc1, false, false);
    acc2 = __builtin_amdgcn_wmma_f32_16x16x32_f16(false, a, false, b2, (short)0, acc2, false, false);
    acc3 = __builtin_amdgcn_wmma_f32_16x16x32_f16(false, a, false, b3, (short)0, acc3, false, false);
  }
  // C/D layout: element r -> row m0 + r + 8*g, col = tile_n0 + lm
  #pragma unroll
  for (int r = 0; r < 8; ++r) {
    size_t row = (size_t)(m0 + r + 8 * g);
    float* yr = Y + row * Nc + n0 + lm;
    yr[0]  = acc0[r] + bias[n0 + lm];
    yr[16] = acc1[r] + bias[n0 + 16 + lm];
    yr[32] = acc2[r] + bias[n0 + 32 + lm];
    yr[48] = acc3[r] + bias[n0 + 48 + lm];
  }
}

// ---------------- edge pass 1: attention logits + segment max ----------------
template <int F, int NH>
__global__ void edge_logits_kernel(const float* __restrict__ xl, const float* __restrict__ xr,
                                   const int* __restrict__ ei,
                                   const float* __restrict__ eattr, const float* __restrict__ loopattr,
                                   const float* __restrict__ We, const float* __restrict__ att,
                                   float* __restrict__ logitOut, unsigned int* __restrict__ lmaxEnc) {
  constexpr int CH = F / NH;
  constexpr int PER = F / 32;
  __shared__ float sWe[FE * F];
  __shared__ float sAtt[F];
  for (int i = threadIdx.x; i < FE * F; i += blockDim.x) sWe[i] = We[i];
  for (int i = threadIdx.x; i < F; i += blockDim.x) sAtt[i] = att[i];
  __syncthreads();
  int e = blockIdx.x * (blockDim.x >> 5) + (threadIdx.x >> 5);
  int lane = threadIdx.x & 31;
  if (e >= ETOT) return;
  int s, d; const float* ea;
  if (e < NEDGES) { s = ei[e]; d = ei[NEDGES + e]; ea = eattr + (size_t)e * FE; }
  else            { s = e - NEDGES; d = s;         ea = loopattr + (size_t)(e - NEDGES) * FE; }
  float eav[FE];
  #pragma unroll
  for (int j = 0; j < FE; ++j) eav[j] = ea[j];
  float hs[NH];
  #pragma unroll
  for (int h = 0; h < NH; ++h) hs[h] = 0.0f;
  #pragma unroll
  for (int i = 0; i < PER; ++i) {
    int c = lane + 32 * i;
    float ee = 0.0f;
    #pragma unroll
    for (int j = 0; j < FE; ++j) ee = fmaf(eav[j], sWe[j * F + c], ee);
    float m = xl[(size_t)s * F + c] + xr[(size_t)d * F + c] + ee;
    m = (m > 0.0f) ? m : SLOPE * m;
    hs[(32 * i) / CH] += sAtt[c] * m;   // head index is compile-time per unrolled i
  }
  #pragma unroll
  for (int h = 0; h < NH; ++h) {
    float v = hs[h];
    #pragma unroll
    for (int off = 16; off >= 1; off >>= 1) v += __shfl_xor(v, off, 32);
    if (lane == 0) {
      logitOut[(size_t)e * NH + h] = v;
      atomicMax(&lmaxEnc[(size_t)d * NH + h], fenc(v));
    }
  }
}

// ---------------- edge pass 2: exp(logit - max), segment sum ----------------
template <int NH>
__global__ void edge_softmax_kernel(const int* __restrict__ ei, float* __restrict__ logits,
                                    const unsigned int* __restrict__ lmaxEnc, float* __restrict__ den) {
  int idx = blockIdx.x * blockDim.x + threadIdx.x;
  if (idx >= ETOT * NH) return;
  int e = idx / NH, h = idx % NH;
  int d = (e < NEDGES) ? ei[NEDGES + e] : (e - NEDGES);
  float num = expf(logits[idx] - fdec(lmaxEnc[(size_t)d * NH + h]));
  logits[idx] = num;
  atomicAdd(&den[(size_t)d * NH + h], num);
}

// ---------------- edge pass 3: out[dst] += alpha * xl[src] ----------------
template <int F, int NH>
__global__ void edge_aggregate_kernel(const float* __restrict__ xl, const int* __restrict__ ei,
                                      const float* __restrict__ num, const float* __restrict__ den,
                                      float* __restrict__ outAccum) {
  constexpr int CH = F / NH;
  constexpr int PER = F / 32;
  int e = blockIdx.x * (blockDim.x >> 5) + (threadIdx.x >> 5);
  int lane = threadIdx.x & 31;
  if (e >= ETOT) return;
  int s, d;
  if (e < NEDGES) { s = ei[e]; d = ei[NEDGES + e]; }
  else            { s = e - NEDGES; d = s; }
  float alpha[NH];
  #pragma unroll
  for (int h = 0; h < NH; ++h)
    alpha[h] = num[(size_t)e * NH + h] / den[(size_t)d * NH + h];
  #pragma unroll
  for (int i = 0; i < PER; ++i) {
    int c = lane + 32 * i;
    atomicAdd(&outAccum[(size_t)d * F + c], alpha[(32 * i) / CH] * xl[(size_t)s * F + c]);
  }
}

// ---------------- batchnorm stats / apply + residual + ELU ----------------
template <int F>
__global__ void bn_sums_kernel(const float* __restrict__ X, float* __restrict__ sum, float* __restrict__ sq) {
  int c = threadIdx.x;               // blockDim.x == F
  int r0 = blockIdx.x * 256;
  int r1 = min(r0 + 256, NNODES);
  float s = 0.0f, s2 = 0.0f;
  for (int r = r0; r < r1; ++r) { float v = X[(size_t)r * F + c]; s += v; s2 += v * v; }
  atomicAdd(&sum[c], s);
  atomicAdd(&sq[c], s2);
}
template <int F>
__global__ void bn_apply_elu_kernel(const float* __restrict__ X, const float* __restrict__ sum,
                                    const float* __restrict__ sq, const float* __restrict__ gma,
                                    const float* __restrict__ bta, const float* __restrict__ skip,
                                    float* __restrict__ out) {
  int idx = blockIdx.x * blockDim.x + threadIdx.x;
  if (idx >= NNODES * F) return;
  int c = idx % F;
  float mu  = sum[c] * (1.0f / NNODES);
  float var = sq[c]  * (1.0f / NNODES) - mu * mu;
  float v = gma[c] * (X[idx] - mu) * rsqrtf(var + BN_EPS) + bta[c] + skip[idx];
  out[idx] = (v > 0.0f) ? v : (expf(v) - 1.0f);
}

// ---------------- global mean+max pooling over graphs ----------------
__global__ void pool_accum_kernel(const float* __restrict__ h2, const int* __restrict__ batch,
                                  float* __restrict__ psum, unsigned int* __restrict__ pmax,
                                  float* __restrict__ pcnt) {
  int idx = blockIdx.x * blockDim.x + threadIdx.x;
  if (idx >= NNODES * F2) return;
  int n = idx / F2, c = idx % F2;
  int g = batch[n];
  float v = h2[idx];
  atomicAdd(&psum[g * F2 + c], v);
  atomicMax(&pmax[g * F2 + c], fenc(v));
  if (c == 0) atomicAdd(&pcnt[g], 1.0f);
}
__global__ void pool_final_kernel(const float* __restrict__ psum, const unsigned int* __restrict__ pmax,
                                  const float* __restrict__ pcnt, float* __restrict__ out) {
  int idx = blockIdx.x * blockDim.x + threadIdx.x;
  if (idx >= GG * F2) return;
  int g = idx / F2, c = idx % F2;
  out[g * (2 * F2) + c]      = psum[idx] / fmaxf(pcnt[g], 1.0f);
  out[g * (2 * F2) + F2 + c] = fdec(pmax[idx]);
}

// ---------------- host launch ----------------
static inline int cdiv(long long a, long long b) { return (int)((a + b - 1) / b); }

extern "C" void kernel_launch(void* const* d_in, const int* in_sizes, int n_in,
                              void* d_out, int out_size, void* d_ws, size_t ws_size,
                              hipStream_t stream) {
  (void)in_sizes; (void)n_in; (void)out_size; (void)ws_size;
  // inputs in setup_inputs() dict order, params flattened in insertion order
  const float* x       = (const float*)d_in[0];
  const int*   ei      = (const int*)  d_in[1];   // [2, E]
  const float* eattr   = (const float*)d_in[2];   // [E, 16]
  const int*   batch   = (const int*)  d_in[3];   // [N]
  const float* skip1W  = (const float*)d_in[4];
  const float* skip1b  = (const float*)d_in[5];
  const float* c1Wl    = (const float*)d_in[6];
  const float* c1bl    = (const float*)d_in[7];
  const float* c1Wr    = (const float*)d_in[8];
  const float* c1br    = (const float*)d_in[9];
  const float* c1We    = (const float*)d_in[10];
  const float* c1att   = (const float*)d_in[11];
  /* c1bias d_in[12]: irrelevant pre-BN (BN is shift invariant) */
  const float* bn1g    = (const float*)d_in[13];
  const float* bn1b    = (const float*)d_in[14];
  const float* skip2W  = (const float*)d_in[15];
  const float* skip2b  = (const float*)d_in[16];
  const float* c2Wl    = (const float*)d_in[17];
  const float* c2bl    = (const float*)d_in[18];
  const float* c2Wr    = (const float*)d_in[19];
  const float* c2br    = (const float*)d_in[20];
  const float* c2We    = (const float*)d_in[21];
  const float* c2att   = (const float*)d_in[22];
  const float* bn2g    = (const float*)d_in[24];
  const float* bn2b    = (const float*)d_in[25];

  float* ws = (float*)d_ws;
  float* out = (float*)d_out;

  // 0) zero workspace (accumulators + encoded-max sentinels); ~207MB ≈ 9us at HBM peak
  fill_zero_kernel<<<2048, 256, 0, stream>>>(ws, WS_FLOATS);

  // 1) self-loop edge attrs (mean of incoming)
  loop_accum_kernel<<<cdiv((long long)NEDGES * FE, 256), 256, 0, stream>>>(
      ei, eattr, ws + O_DEG, ws + O_LOOP);
  loop_final_kernel<<<cdiv((long long)NNODES * FE, 256), 256, 0, stream>>>(ws + O_DEG, ws + O_LOOP);

  // 2) block-1 node transforms via WMMA (16x64 tile per wave, 4 wmma/iter)
  {
    int waves = (NNODES / 16) * (F1 / 64);
    int blocks = cdiv(waves, 8);
    gemm_wmma_kernel<<<blocks, 256, 0, stream>>>(x, skip1W, skip1b, ws + O_XP1, NNODES, FIN, F1);
    gemm_wmma_kernel<<<blocks, 256, 0, stream>>>(x, c1Wl,  c1bl,  ws + O_XL1, NNODES, FIN, F1);
    gemm_wmma_kernel<<<blocks, 256, 0, stream>>>(x, c1Wr,  c1br,  ws + O_XR1, NNODES, FIN, F1);
  }

  // 3) block-1 edge attention (3-pass segment softmax + aggregate)
  edge_logits_kernel<F1, NH1><<<cdiv(ETOT, 8), 256, 0, stream>>>(
      ws + O_XL1, ws + O_XR1, ei, eattr, ws + O_LOOP, c1We, c1att,
      ws + O_LOG1, (unsigned int*)(ws + O_LMAX1));
  edge_softmax_kernel<NH1><<<cdiv((long long)ETOT * NH1, 256), 256, 0, stream>>>(
      ei, ws + O_LOG1, (const unsigned int*)(ws + O_LMAX1), ws + O_DEN1);
  edge_aggregate_kernel<F1, NH1><<<cdiv(ETOT, 8), 256, 0, stream>>>(
      ws + O_XL1, ei, ws + O_LOG1, ws + O_DEN1, ws + O_OUT1);

  // 4) BN1 + residual + ELU -> h1
  bn_sums_kernel<F1><<<cdiv(NNODES, 256), F1, 0, stream>>>(ws + O_OUT1, ws + O_BS1, ws + O_BQ1);
  bn_apply_elu_kernel<F1><<<cdiv((long long)NNODES * F1, 256), 256, 0, stream>>>(
      ws + O_OUT1, ws + O_BS1, ws + O_BQ1, bn1g, bn1b, ws + O_XP1, ws + O_H1);

  // 5) block-2 node transforms via WMMA (K=128, Nc=64)
  {
    int waves = (NNODES / 16) * (F2 / 64);
    int blocks = cdiv(waves, 8);
    gemm_wmma_kernel<<<blocks, 256, 0, stream>>>(ws + O_H1, skip2W, skip2b, ws + O_XP2, NNODES, F1, F2);
    gemm_wmma_kernel<<<blocks, 256, 0, stream>>>(ws + O_H1, c2Wl,  c2bl,  ws + O_XL2, NNODES, F1, F2);
    gemm_wmma_kernel<<<blocks, 256, 0, stream>>>(ws + O_H1, c2Wr,  c2br,  ws + O_XR2, NNODES, F1, F2);
  }

  // 6) block-2 edge attention
  edge_logits_kernel<F2, 1><<<cdiv(ETOT, 8), 256, 0, stream>>>(
      ws + O_XL2, ws + O_XR2, ei, eattr, ws + O_LOOP, c2We, c2att,
      ws + O_LOG2, (unsigned int*)(ws + O_LMAX2));
  edge_softmax_kernel<1><<<cdiv(ETOT, 256), 256, 0, stream>>>(
      ei, ws + O_LOG2, (const unsigned int*)(ws + O_LMAX2), ws + O_DEN2);
  edge_aggregate_kernel<F2, 1><<<cdiv(ETOT, 8), 256, 0, stream>>>(
      ws + O_XL2, ei, ws + O_LOG2, ws + O_DEN2, ws + O_OUT2);

  // 7) BN2 + residual + ELU -> h2
  bn_sums_kernel<F2><<<cdiv(NNODES, 256), F2, 0, stream>>>(ws + O_OUT2, ws + O_BS2, ws + O_BQ2);
  bn_apply_elu_kernel<F2><<<cdiv((long long)NNODES * F2, 256), 256, 0, stream>>>(
      ws + O_OUT2, ws + O_BS2, ws + O_BQ2, bn2g, bn2b, ws + O_XP2, ws + O_H2);

  // 8) global mean+max pooling -> out [8, 128]
  pool_accum_kernel<<<cdiv((long long)NNODES * F2, 256), 256, 0, stream>>>(
      ws + O_H2, batch, ws + O_PSUM, (unsigned int*)(ws + O_PMAX), ws + O_PCNT);
  pool_final_kernel<<<cdiv(GG * F2, 256), 256, 0, stream>>>(
      ws + O_PSUM, (const unsigned int*)(ws + O_PMAX), ws + O_PCNT, out);
}
